// LogMM_19490561589867
// MI455X (gfx1250) — matrix operand order repaired and checked by reading, not compiled, not run on hardware
//
#include <hip/hip_runtime.h>
#include <cstdint>

// ---------------------------------------------------------------------------
// LogMM on MI455X (gfx1250): out = log(max(x @ matrix, FLT_TINY))
//   M=16384, N=1024, K=1024, fp32 in/out.  bf16x3 split GEMM on
//   V_WMMA_F32_16X16X32_BF16, TDM (tensor_load_to_lds) double-buffered LDS
//   staging, fused log epilogue.
// ---------------------------------------------------------------------------

typedef __attribute__((ext_vector_type(16))) __bf16 v16bf;
typedef __attribute__((ext_vector_type(8)))  __bf16 v8bf;
typedef __attribute__((ext_vector_type(8)))  float  v8f;
typedef __attribute__((ext_vector_type(4)))  unsigned int u32x4;
typedef __attribute__((ext_vector_type(8)))  int i32x8;
typedef __attribute__((ext_vector_type(4)))  int i32x4;

#define MDIM 16384
#define NDIM 1024
#define KDIM 1024

// LDS tile geometry: 128 rows x 32 bf16, padded to 80B row stride
// (TDM pad: 4 DWORDs every 16 DWORDs) -> conflict-free ds_load_b128.
#define TILE_BYTES   (128 * 80)          // 10240
#define STAGE_BYTES  (4 * TILE_BYTES)    // Ahi|Alo|Bhi|Blo = 40960
#define LDS_BYTES    (2 * STAGE_BYTES)   // double buffered = 81920

#if defined(__AMDGCN__) && __has_builtin(__builtin_amdgcn_tensor_load_to_lds) && \
    __has_builtin(__builtin_amdgcn_s_wait_tensorcnt)
#define LOGMM_USE_TDM 1
#endif

__device__ __forceinline__ unsigned short f2bf_rne(float f) {
  unsigned u = __float_as_uint(f);
  u += 0x7FFFu + ((u >> 16) & 1u);       // round to nearest even
  return (unsigned short)(u >> 16);
}
__device__ __forceinline__ float bf2f(unsigned short h) {
  return __uint_as_float(((unsigned)h) << 16);
}

// --- Pre-pass 1: split x (row-major [M][K] fp32) into hi/lo bf16 ------------
__global__ __launch_bounds__(256) void split_a_kernel(
    const float* __restrict__ a,
    unsigned short* __restrict__ hi,
    unsigned short* __restrict__ lo)
{
  size_t i4 = ((size_t)blockIdx.x * 256 + threadIdx.x) * 4;
  float4 v = *(const float4*)(a + i4);
  float f[4] = {v.x, v.y, v.z, v.w};
  unsigned short hb[4], lb[4];
#pragma unroll
  for (int j = 0; j < 4; ++j) {
    hb[j] = f2bf_rne(f[j]);
    lb[j] = f2bf_rne(f[j] - bf2f(hb[j]));
  }
  ushort4 h; h.x = hb[0]; h.y = hb[1]; h.z = hb[2]; h.w = hb[3];
  ushort4 l; l.x = lb[0]; l.y = lb[1]; l.z = lb[2]; l.w = lb[3];
  *(ushort4*)(hi + i4) = h;
  *(ushort4*)(lo + i4) = l;
}

// --- Pre-pass 2: transpose+split matrix [K][N] -> Bt hi/lo [N][K] bf16 ------
__global__ __launch_bounds__(256) void split_bt_kernel(
    const float* __restrict__ b,
    unsigned short* __restrict__ hi,
    unsigned short* __restrict__ lo)
{
  size_t t = (size_t)blockIdx.x * 256 + threadIdx.x;  // over N*K, k fastest
  int n = (int)(t >> 10);
  int k = (int)(t & (KDIM - 1));
  float v = b[(size_t)k * NDIM + n];
  unsigned short hb = f2bf_rne(v);
  hi[t] = hb;
  lo[t] = f2bf_rne(v - bf2f(hb));
}

#ifdef LOGMM_USE_TDM
// Issue one TDM 2D tile load: 128 lines x 32 bf16, line stride 1024 elements,
// LDS padding 4 DWORDs per 16 DWORDs (-> 80B LDS row stride).
__device__ __forceinline__ void tdm_load_tile_2d(unsigned lds_addr,
                                                 const void* gptr,
                                                 i32x8 g1) {
  unsigned long long ga = (unsigned long long)(size_t)gptr;
  u32x4 g0;
  g0.x = 1u;                                              // count=1, user D#
  g0.y = lds_addr;                                        // lds_addr[31:0]
  g0.z = (unsigned)(ga & 0xFFFFFFFFull);                  // global_addr lo
  g0.w = (unsigned)((ga >> 32) & 0x01FFFFFFull) | (2u << 30);  // hi | type=2
  i32x4 z4 = {0, 0, 0, 0};
#if __has_include(<hip/amd_detail/amd_gfx1250_TDM.h>)
  i32x8 z8 = {0, 0, 0, 0, 0, 0, 0, 0};
  __builtin_amdgcn_tensor_load_to_lds(g0, g1, z4, z4, z8, 0);   // 6-arg lane
#else
  __builtin_amdgcn_tensor_load_to_lds(g0, g1, z4, z4, 0);       // 5-arg lane
#endif
}
#endif

// --- Main GEMM + log epilogue ----------------------------------------------
// 256 threads = 8 wave32 (4 M-waves x 2 N-waves); WG tile 128x128; wave tile
// 32x64 = 2x4 WMMA tiles; K stepped by 32; bf16x3 -> 24 WMMAs / K-step / wave.
__global__ __launch_bounds__(256) void gemm_log_kernel(
    const __bf16* __restrict__ Ahi, const __bf16* __restrict__ Alo,
    const __bf16* __restrict__ Bhi, const __bf16* __restrict__ Blo,
    float* __restrict__ out)
{
  const int lane    = threadIdx.x & 31;
  const int wave    = threadIdx.x >> 5;
  const int wm      = wave & 3;    // 4 waves along M
  const int wn      = wave >> 2;   // 2 waves along N
  const int lane_lo = lane & 15;
  const int lane_hi = lane >> 4;

  const int row_base = blockIdx.y * 128 + wm * 32;
  const int col_base = blockIdx.x * 128 + wn * 64;

  v8f zero = {0.f, 0.f, 0.f, 0.f, 0.f, 0.f, 0.f, 0.f};
  v8f acc[2][4];
#pragma unroll
  for (int mt = 0; mt < 2; ++mt)
#pragma unroll
    for (int nt = 0; nt < 4; ++nt)
      acc[mt][nt] = zero;

#ifdef LOGMM_USE_TDM
  // ---------------- TDM + LDS double-buffered path ----------------
  __shared__ alignas(128) unsigned char smem[LDS_BYTES];
  const unsigned ldsBase = (unsigned)(size_t)(void*)smem;

  // D# group 1 (identical for all 4 tiles):
  //   data_size=1 (2B), pad_enable, pad_interval=3 (16 DW), pad_amount=3 (4 DW)
  //   tensor_dim0=32, tensor_dim1=128, tile_dim0=32, tile_dim1=128,
  //   tile_dim2=0, tensor_dim0_stride=1024, tensor_dim1_stride=0.
  i32x8 g1;
  g1[0] = (int)((1u << 16) | (1u << 20) | (3u << 22) | (3u << 25));
  g1[1] = (int)(32u << 16);            // tensor_dim0[15:0] @ bits 63:48
  g1[2] = (int)(128u << 16);           // tensor_dim0[31:16]=0 | tensor_dim1[15:0]
  g1[3] = (int)(32u << 16);            // tensor_dim1[31:16]=0 | tile_dim0
  g1[4] = (int)(128u);                 // tile_dim1 | tile_dim2<<16
  g1[5] = (int)(1024);                 // tensor_dim0_stride[31:0]
  g1[6] = 0;                           // stride0[47:32] | stride1[15:0]<<16
  g1[7] = 0;                           // stride1[47:16]

  const size_t aTileOff = (size_t)blockIdx.y * 128 * KDIM;
  const size_t bTileOff = (size_t)blockIdx.x * 128 * KDIM;

  auto issue_stage = [&](int k0, int buf) {
    unsigned b = ldsBase + (unsigned)buf * STAGE_BYTES;
    tdm_load_tile_2d(b + 0u * TILE_BYTES, Ahi + aTileOff + k0, g1);
    tdm_load_tile_2d(b + 1u * TILE_BYTES, Alo + aTileOff + k0, g1);
    tdm_load_tile_2d(b + 2u * TILE_BYTES, Bhi + bTileOff + k0, g1);
    tdm_load_tile_2d(b + 3u * TILE_BYTES, Blo + bTileOff + k0, g1);
  };

  if (wave == 0) issue_stage(0, 0);

  for (int s = 0; s < KDIM / 32; ++s) {
    const int p = s & 1;
    if (wave == 0) __builtin_amdgcn_s_wait_tensorcnt(0);  // stage s landed
    __syncthreads();
    if (wave == 0 && s + 1 < KDIM / 32)
      issue_stage((s + 1) * 32, p ^ 1);                   // overlap DMA s+1

    const unsigned buf = (unsigned)p * STAGE_BYTES;

    v16bf ah[2], al[2];
#pragma unroll
    for (int mt = 0; mt < 2; ++mt) {
      // A 16x32 bf16 layout: lanes 0-15 hold K[0..7],K[16..23] of row lane&15;
      // lanes 16-31 hold K[8..15],K[24..31].
      unsigned r = (unsigned)(wm * 32 + mt * 16 + lane_lo) * 80u +
                   (unsigned)lane_hi * 16u;
      const unsigned char* ph = smem + buf + 0u * TILE_BYTES + r;
      const unsigned char* pl = smem + buf + 1u * TILE_BYTES + r;
      v8bf h0 = *(const v8bf*)ph;
      v8bf h1 = *(const v8bf*)(ph + 32);
      v8bf l0 = *(const v8bf*)pl;
      v8bf l1 = *(const v8bf*)(pl + 32);
      ah[mt] = __builtin_shufflevector(h0, h1, 0,1,2,3,4,5,6,7,8,9,10,11,12,13,14,15);
      al[mt] = __builtin_shufflevector(l0, l1, 0,1,2,3,4,5,6,7,8,9,10,11,12,13,14,15);
    }

    v16bf bh[4], bl[4];
#pragma unroll
    for (int nt = 0; nt < 4; ++nt) {
      // B 32x16 bf16 layout: lanes 0-15 hold K[0..7],K[16..23] of col lane&15;
      // lanes 16-31 hold K[8..15],K[24..31].
      unsigned c = (unsigned)(wn * 64 + nt * 16 + lane_lo) * 80u +
                   (unsigned)lane_hi * 16u;
      const unsigned char* ph = smem + buf + 2u * TILE_BYTES + c;
      const unsigned char* pl = smem + buf + 3u * TILE_BYTES + c;
      v8bf h0 = *(const v8bf*)ph;
      v8bf h1 = *(const v8bf*)(ph + 32);
      v8bf l0 = *(const v8bf*)pl;
      v8bf l1 = *(const v8bf*)(pl + 32);
      bh[nt] = __builtin_shufflevector(h0, h1, 0,1,2,3,4,5,6,7,8,9,10,11,12,13,14,15);
      bl[nt] = __builtin_shufflevector(l0, l1, 0,1,2,3,4,5,6,7,8,9,10,11,12,13,14,15);
    }

#pragma unroll
    for (int mt = 0; mt < 2; ++mt)
#pragma unroll
      for (int nt = 0; nt < 4; ++nt) {
        acc[mt][nt] = __builtin_amdgcn_wmma_f32_16x16x32_bf16(
            false, ah[mt], false, bh[nt], (short)0, acc[mt][nt], false, false);
        acc[mt][nt] = __builtin_amdgcn_wmma_f32_16x16x32_bf16(
            false, ah[mt], false, bl[nt], (short)0, acc[mt][nt], false, false);
        acc[mt][nt] = __builtin_amdgcn_wmma_f32_16x16x32_bf16(
            false, al[mt], false, bh[nt], (short)0, acc[mt][nt], false, false);
      }
    __syncthreads();   // buffer p free for stage s+2
  }
#else
  // ---------------- direct-global fallback (also host-parse body) -----------
  size_t aOff[2];
#pragma unroll
  for (int mt = 0; mt < 2; ++mt)
    aOff[mt] = (size_t)(row_base + mt * 16 + lane_lo) * KDIM + lane_hi * 16;
  size_t bOff[4];
#pragma unroll
  for (int nt = 0; nt < 4; ++nt)
    bOff[nt] = (size_t)(col_base + nt * 16 + lane_lo) * KDIM + lane_hi * 16;

  for (int k0 = 0; k0 < KDIM; k0 += 32) {
    v16bf ah[2], al[2];
#pragma unroll
    for (int mt = 0; mt < 2; ++mt) {
      const __bf16* ph = Ahi + aOff[mt] + k0;
      const __bf16* pl = Alo + aOff[mt] + k0;
      v8bf h0 = *(const v8bf*)ph;
      v8bf h1 = *(const v8bf*)(ph + 8);
      v8bf l0 = *(const v8bf*)pl;
      v8bf l1 = *(const v8bf*)(pl + 8);
      ah[mt] = __builtin_shufflevector(h0, h1, 0,1,2,3,4,5,6,7,8,9,10,11,12,13,14,15);
      al[mt] = __builtin_shufflevector(l0, l1, 0,1,2,3,4,5,6,7,8,9,10,11,12,13,14,15);
    }
    v16bf bh[4], bl[4];
#pragma unroll
    for (int nt = 0; nt < 4; ++nt) {
      bh[nt] = *(const v16bf*)(Bhi + bOff[nt] + k0);
      bl[nt] = *(const v16bf*)(Blo + bOff[nt] + k0);
    }
#pragma unroll
    for (int mt = 0; mt < 2; ++mt)
#pragma unroll
      for (int nt = 0; nt < 4; ++nt) {
        acc[mt][nt] = __builtin_amdgcn_wmma_f32_16x16x32_bf16(
            false, ah[mt], false, bh[nt], (short)0, acc[mt][nt], false, false);
        acc[mt][nt] = __builtin_amdgcn_wmma_f32_16x16x32_bf16(
            false, ah[mt], false, bl[nt], (short)0, acc[mt][nt], false, false);
        acc[mt][nt] = __builtin_amdgcn_wmma_f32_16x16x32_bf16(
            false, al[mt], false, bh[nt], (short)0, acc[mt][nt], false, false);
      }
  }
#endif

  // Epilogue: out = log(max(acc, FLT_TINY)).
  // C/D layout: VGPR r -> row (r + 8*lane_hi), col = lane&15.
  const float tiny = 1.17549435e-38f;
#pragma unroll
  for (int mt = 0; mt < 2; ++mt) {
#pragma unroll
    for (int nt = 0; nt < 4; ++nt) {
      const int r0 = row_base + mt * 16 + lane_hi * 8;
      const int c  = col_base + nt * 16 + lane_lo;
#pragma unroll
      for (int r = 0; r < 8; ++r) {
        float v = acc[mt][nt][r];
        out[(size_t)(r0 + r) * NDIM + c] = __logf(fmaxf(v, tiny));
      }
    }
  }
}

// ---------------------------------------------------------------------------
extern "C" void kernel_launch(void* const* d_in, const int* in_sizes, int n_in,
                              void* d_out, int out_size, void* d_ws, size_t ws_size,
                              hipStream_t stream) {
  (void)in_sizes; (void)n_in; (void)out_size; (void)ws_size;
  const float* x   = (const float*)d_in[0];   // [8,2048,1024] -> [16384,1024]
  const float* mat = (const float*)d_in[1];   // [1024,1024]
  float* out = (float*)d_out;

  // Workspace layout (68 MB): A_hi | A_lo | Bt_hi | Bt_lo  (bf16 each)
  unsigned short* Ahi = (unsigned short*)d_ws;
  unsigned short* Alo = Ahi + (size_t)MDIM * KDIM;
  unsigned short* Bhi = Alo + (size_t)MDIM * KDIM;
  unsigned short* Blo = Bhi + (size_t)NDIM * KDIM;

  {
    size_t n4 = (size_t)MDIM * KDIM / 4;   // 4 elems per thread
    split_a_kernel<<<dim3((unsigned)(n4 / 256)), dim3(256), 0, stream>>>(x, Ahi, Alo);
  }
  {
    size_t n = (size_t)NDIM * KDIM;
    split_bt_kernel<<<dim3((unsigned)(n / 256)), dim3(256), 0, stream>>>(mat, Bhi, Blo);
  }
  // N-blocks fastest so the 8 WGs sharing an A stripe are L2-coresident.
  gemm_log_kernel<<<dim3(NDIM / 128, MDIM / 128), dim3(256), 0, stream>>>(
      (const __bf16*)Ahi, (const __bf16*)Alo,
      (const __bf16*)Bhi, (const __bf16*)Blo, out);
}
